// GatedGCNDecoder_21887153340950
// MI455X (gfx1250) — matrix operand order repaired and checked by reading, not compiled, not run on hardware
//
#include <hip/hip_runtime.h>
#include <hip/hip_bf16.h>

#define NN      10000
#define DEG     32
#define H       256
#define OUTC    128
#define LAYERS  3
#define LN_EPS  1e-5f
#define KT      (H / 32)      // 8 K-tiles of 32

typedef __attribute__((ext_vector_type(16))) __bf16 v16bf;
typedef __attribute__((ext_vector_type(8)))  float  v8f;

__device__ __forceinline__ __bf16 f2bf(float f) { return (__bf16)f; }

// A-fragment K index for 16-bit 16x32 A: lane half h, element e -> K = 16h + e (matches B packing)
__device__ __forceinline__ int kidxA(int e, int h) {
    return 16 * h + e;
}

// ---------------------------------------------------------------------------
// Pack activations [M,H] fp32 (row-major) -> bf16 WMMA A-fragments.
// Fragment (mb,kb) stored at ((mb*KT+kb)*32+lane)*16, 32B per lane.
// ---------------------------------------------------------------------------
__global__ void pack_a(const float* __restrict__ X, __bf16* __restrict__ A, int Mtiles) {
    int frag = blockIdx.x * 8 + (threadIdx.x >> 5);
    int lane = threadIdx.x & 31;
    if (frag >= Mtiles * KT) return;
    int kb = frag & (KT - 1);
    int mb = frag / KT;
    int m  = mb * 16 + (lane & 15);
    int h2 = lane >> 4;
    __bf16* dst = A + ((size_t)frag * 32 + lane) * 16;
    const float* src = X + (size_t)m * H + kb * 32;
#pragma unroll
    for (int e = 0; e < 16; ++e) dst[e] = f2bf(src[kidxA(e, h2)]);
}

// ---------------------------------------------------------------------------
// Pack weight [K=256, Ncols] fp32 (row-major) -> bf16 WMMA B-fragments.
// Fragment (nb,kb) stored at ((nb*KT+kb)*32+lane)*16.
// B packing order used here (same as A): lane half h, element e -> K = 16h + e, N = lane&15.
// ---------------------------------------------------------------------------
__global__ void pack_w(const float* __restrict__ W, __bf16* __restrict__ P, int Ncols) {
    int frag = blockIdx.x * 8 + (threadIdx.x >> 5);
    int lane = threadIdx.x & 31;
    int Ntiles = Ncols / 16;
    if (frag >= KT * Ntiles) return;
    int kb = frag % KT;
    int nb = frag / KT;
    int n  = nb * 16 + (lane & 15);
    int kbase = kb * 32 + 16 * (lane >> 4);
    __bf16* dst = P + ((size_t)frag * 32 + lane) * 16;
#pragma unroll
    for (int e = 0; e < 16; ++e) dst[e] = f2bf(W[(size_t)(kbase + e) * Ncols + n]);
}

// ---------------------------------------------------------------------------
// Build in-neighbor table: nbr[i][0..31] = { j : adj[j*N + i] > 0 }, pad -1.
// One adj scan per launch (~400MB, ~17us at 23.3 TB/s), coalesced over i.
// ---------------------------------------------------------------------------
__global__ void build_nbr(const float* __restrict__ adj, int* __restrict__ nbr) {
    int i = blockIdx.x * blockDim.x + threadIdx.x;
    if (i >= NN) return;
    int c = 0;
    for (int j = 0; j < NN && c < DEG; ++j)
        if (adj[(size_t)j * NN + i] > 0.0f) nbr[i * DEG + c++] = j;
    for (; c < DEG; ++c) nbr[i * DEG + c] = -1;
}

// ---------------------------------------------------------------------------
// bf16 WMMA GEMM: C[M,Ncols] = A[M,256] @ W[256,Ncols] + bias (+optional ReLU)
// Block = 8 waves sharing one 64-col B panel staged in LDS (32KB); each wave
// owns a 16-row A panel preloaded into registers. The 4 B fragments are
// register double-buffered across K-steps so ds_loads for kb+1 overlap the
// 4 WMMAs of kb instead of stalling on s_wait_dscnt at point of use.
// ---------------------------------------------------------------------------
__global__ void __launch_bounds__(256)
gemm_bf16(const __bf16* __restrict__ A, const __bf16* __restrict__ B,
          const float* __restrict__ bias, float* __restrict__ C,
          int Mtiles, int Ncols, int relu) {
    __shared__ __bf16 Bsh[4 * KT * 512];              // 4 n-tiles x 8 k-tiles x 1KB = 32KB

    int tid = threadIdx.x;
    // ---- cooperative copy of the block's contiguous 32KB B panel ----
    {
        const float4* src = (const float4*)(B + (size_t)(blockIdx.y * 4) * KT * 512);
        float4* dst = (float4*)Bsh;
#pragma unroll
        for (int t = 0; t < 8; ++t)                   // 2048 float4 / 256 threads
            dst[tid + 256 * t] = src[tid + 256 * t];
    }
    __syncthreads();

    int wave = tid >> 5;
    int lane = tid & 31;
    int mtile = blockIdx.x * 8 + wave;
    if (mtile >= Mtiles) return;                      // wave-uniform; no barriers below

    // ---- preload all A fragments for this wave's 16 rows ----
    const __bf16* ap = A + ((size_t)mtile * KT * 32 + lane) * 16;
    v16bf a[KT];
#pragma unroll
    for (int kb = 0; kb < KT; ++kb)
        a[kb] = *(const v16bf*)(ap + (size_t)kb * 512);

    const __bf16* bsh = Bsh + lane * 16;
    const size_t nstride = (size_t)KT * 512;          // elems between 16-col panels

    v8f acc0 = {}, acc1 = {}, acc2 = {}, acc3 = {};

    // ---- software-pipelined K loop: double-buffer B fragments ----
    v16bf b0 = *(const v16bf*)(bsh);
    v16bf b1 = *(const v16bf*)(bsh + nstride);
    v16bf b2 = *(const v16bf*)(bsh + 2 * nstride);
    v16bf b3 = *(const v16bf*)(bsh + 3 * nstride);
#pragma unroll
    for (int kb = 0; kb < KT; ++kb) {
        v16bf c0 = b0, c1 = b1, c2 = b2, c3 = b3;
        if (kb + 1 < KT) {
            const __bf16* bp = bsh + (size_t)(kb + 1) * 512;
            b0 = *(const v16bf*)(bp);
            b1 = *(const v16bf*)(bp + nstride);
            b2 = *(const v16bf*)(bp + 2 * nstride);
            b3 = *(const v16bf*)(bp + 3 * nstride);
        }
        acc0 = __builtin_amdgcn_wmma_f32_16x16x32_bf16(false, a[kb], false, c0, (short)0, acc0, false, false);
        acc1 = __builtin_amdgcn_wmma_f32_16x16x32_bf16(false, a[kb], false, c1, (short)0, acc1, false, false);
        acc2 = __builtin_amdgcn_wmma_f32_16x16x32_bf16(false, a[kb], false, c2, (short)0, acc2, false, false);
        acc3 = __builtin_amdgcn_wmma_f32_16x16x32_bf16(false, a[kb], false, c3, (short)0, acc3, false, false);
    }

    // C/D layout: VGPR r -> row 8*(lane>>4)+r, col = lane&15 within 16x16 tile
    int rbase = mtile * 16 + 8 * (lane >> 4);
    int nlo   = lane & 15;
    int nb0   = blockIdx.y * 4;
    v8f accs[4] = {acc0, acc1, acc2, acc3};
#pragma unroll
    for (int j = 0; j < 4; ++j) {
        int col = (nb0 + j) * 16 + nlo;
        float bv = bias ? bias[col] : 0.0f;
#pragma unroll
        for (int r = 0; r < 8; ++r) {
            float val = accs[j][r] + bv;
            if (relu) val = fmaxf(val, 0.0f);
            C[(size_t)(rbase + r) * Ncols + col] = val;
        }
    }
}

// ---------------------------------------------------------------------------
// Gated aggregation: x[i] = sum_j sigmoid(k_i + q_j) * v_j  + (x@Wskip + conv_b)
// One block per node, one thread per hidden dim; q/v gathers hit L2.
// ---------------------------------------------------------------------------
__global__ void agg_kernel(const float* __restrict__ kbuf, const float* __restrict__ qbuf,
                           const float* __restrict__ vbuf, const float* __restrict__ sbuf,
                           const int* __restrict__ nbr, float* __restrict__ xout) {
    int i = blockIdx.x;
    int d = threadIdx.x;
    float ki  = kbuf[(size_t)i * H + d];
    float acc = 0.0f;
#pragma unroll 4
    for (int n = 0; n < DEG; ++n) {
        int j = nbr[i * DEG + n];
        if (j < 0) continue;
        float g = 1.0f / (1.0f + __expf(-(ki + qbuf[(size_t)j * H + d])));
        acc += g * vbuf[(size_t)j * H + d];
    }
    xout[(size_t)i * H + d] = acc + sbuf[(size_t)i * H + d];
}

// ---------------------------------------------------------------------------
// Fused LayerNorm + ReLU: one wave32 per row, 8 elems per lane, shfl reduce.
// ---------------------------------------------------------------------------
__global__ void ln_relu(const float* __restrict__ hbuf, const float* __restrict__ gam,
                        const float* __restrict__ bet, float* __restrict__ xout) {
    int row  = blockIdx.x * 8 + (threadIdx.x >> 5);
    int lane = threadIdx.x & 31;
    if (row >= NN) return;
    const float* hp = hbuf + (size_t)row * H;
    float v[8];
    float s = 0.0f;
#pragma unroll
    for (int t = 0; t < 8; ++t) { v[t] = hp[lane + 32 * t]; s += v[t]; }
#pragma unroll
    for (int o = 16; o >= 1; o >>= 1) s += __shfl_xor(s, o, 32);
    float mu = s * (1.0f / H);
    float vs = 0.0f;
#pragma unroll
    for (int t = 0; t < 8; ++t) { float dlt = v[t] - mu; vs += dlt * dlt; }
#pragma unroll
    for (int o = 16; o >= 1; o >>= 1) vs += __shfl_xor(vs, o, 32);
    float rstd = rsqrtf(vs * (1.0f / H) + LN_EPS);
#pragma unroll
    for (int t = 0; t < 8; ++t) {
        int col = lane + 32 * t;
        float y = (v[t] - mu) * rstd * gam[col] + bet[col];
        xout[(size_t)row * H + col] = fmaxf(y, 0.0f);
    }
}

// ---------------------------------------------------------------------------
extern "C" void kernel_launch(void* const* d_in, const int* in_sizes, int n_in,
                              void* d_out, int out_size, void* d_ws, size_t ws_size,
                              hipStream_t stream) {
    const float* node_feat = (const float*)d_in[0];
    const float* adj       = (const float*)d_in[1];
    const float* key_w     = (const float*)d_in[2];
    const float* key_b     = (const float*)d_in[3];
    const float* query_w   = (const float*)d_in[4];
    const float* query_b   = (const float*)d_in[5];
    const float* value_w   = (const float*)d_in[6];
    const float* value_b   = (const float*)d_in[7];
    const float* skip_w    = (const float*)d_in[8];
    const float* conv_b    = (const float*)d_in[9];
    const float* mlp_w     = (const float*)d_in[10];
    const float* mlp_b     = (const float*)d_in[11];
    const float* ln_g      = (const float*)d_in[12];
    const float* ln_b      = (const float*)d_in[13];
    const float* f1_w      = (const float*)d_in[14];
    const float* f1_b      = (const float*)d_in[15];
    const float* f2_w      = (const float*)d_in[16];
    const float* f2_b      = (const float*)d_in[17];
    const float* lin_w     = (const float*)d_in[18];
    const float* lin_b     = (const float*)d_in[19];
    float* out = (float*)d_out;

    // ---- workspace carve-up (256B aligned) ----
    char* base = (char*)d_ws;
    size_t off = 0;
    auto alloc = [&](size_t bytes) -> char* {
        char* p = base + off;
        off = (off + bytes + 255) & ~(size_t)255;
        return p;
    };
    const size_t WMAT = (size_t)H * H;          // 65536 elems
    const size_t WOUT = (size_t)H * OUTC;       // 32768 elems
    __bf16* wpack = (__bf16*)alloc((15 * WMAT + WMAT + 2 * WOUT) * sizeof(__bf16));
    int*    nbr   = (int*)  alloc((size_t)NN * DEG * sizeof(int));
    float*  xa    = (float*)alloc((size_t)NN * H * sizeof(float));
    __bf16* apack = (__bf16*)alloc((size_t)NN * H * sizeof(__bf16));
    float*  kbuf  = (float*)alloc((size_t)NN * H * sizeof(float));
    float*  qbuf  = (float*)alloc((size_t)NN * H * sizeof(float));
    float*  vbuf  = (float*)alloc((size_t)NN * H * sizeof(float));
    float*  sbuf  = (float*)alloc((size_t)NN * H * sizeof(float));
    float*  hbuf  = (float*)alloc((size_t)NN * H * sizeof(float));
    float*  tbuf  = (float*)alloc((size_t)NN * H * sizeof(float));

    __bf16* wp_f1  = wpack + 15 * WMAT;
    __bf16* wp_lin = wpack + 16 * WMAT;
    __bf16* wp_f2  = wpack + 16 * WMAT + WOUT;

    const int Mtiles = NN / 16;                 // 625 exact
    dim3 gblk(256);
    dim3 ggemmH((Mtiles + 7) / 8, H / 64);      // 79 x 4
    dim3 ggemmO((Mtiles + 7) / 8, OUTC / 64);   // 79 x 2
    const int PW_H = (KT * (H / 16)) / 8;       // 16 blocks
    const int PW_O = (KT * (OUTC / 16)) / 8;    // 8 blocks
    const int PA   = (Mtiles * KT) / 8;         // 625 blocks

    // ---- one-time per launch: pack weights, build neighbor table ----
    for (int l = 0; l < LAYERS; ++l) {
        pack_w<<<PW_H, gblk, 0, stream>>>(key_w   + (size_t)l * WMAT, wpack + (size_t)(l * 5 + 0) * WMAT, H);
        pack_w<<<PW_H, gblk, 0, stream>>>(query_w + (size_t)l * WMAT, wpack + (size_t)(l * 5 + 1) * WMAT, H);
        pack_w<<<PW_H, gblk, 0, stream>>>(value_w + (size_t)l * WMAT, wpack + (size_t)(l * 5 + 2) * WMAT, H);
        pack_w<<<PW_H, gblk, 0, stream>>>(skip_w  + (size_t)l * WMAT, wpack + (size_t)(l * 5 + 3) * WMAT, H);
        pack_w<<<PW_H, gblk, 0, stream>>>(mlp_w   + (size_t)l * WMAT, wpack + (size_t)(l * 5 + 4) * WMAT, H);
    }
    pack_w<<<PW_H, gblk, 0, stream>>>(f1_w,  wp_f1,  H);
    pack_w<<<PW_O, gblk, 0, stream>>>(lin_w, wp_lin, OUTC);
    pack_w<<<PW_O, gblk, 0, stream>>>(f2_w,  wp_f2,  OUTC);
    build_nbr<<<(NN + 255) / 256, gblk, 0, stream>>>(adj, nbr);

    // ---- layers ----
    const float* cur = node_feat;
    for (int l = 0; l < LAYERS; ++l) {
        pack_a<<<PA, gblk, 0, stream>>>(cur, apack, Mtiles);
        gemm_bf16<<<ggemmH, gblk, 0, stream>>>(apack, wpack + (size_t)(l * 5 + 0) * WMAT,
                                               key_b   + (size_t)l * H, kbuf, Mtiles, H, 0);
        gemm_bf16<<<ggemmH, gblk, 0, stream>>>(apack, wpack + (size_t)(l * 5 + 1) * WMAT,
                                               query_b + (size_t)l * H, qbuf, Mtiles, H, 0);
        gemm_bf16<<<ggemmH, gblk, 0, stream>>>(apack, wpack + (size_t)(l * 5 + 2) * WMAT,
                                               value_b + (size_t)l * H, vbuf, Mtiles, H, 0);
        gemm_bf16<<<ggemmH, gblk, 0, stream>>>(apack, wpack + (size_t)(l * 5 + 3) * WMAT,
                                               conv_b  + (size_t)l * H, sbuf, Mtiles, H, 0);
        agg_kernel<<<NN, gblk, 0, stream>>>(kbuf, qbuf, vbuf, sbuf, nbr, xa);
        pack_a<<<PA, gblk, 0, stream>>>(xa, apack, Mtiles);
        gemm_bf16<<<ggemmH, gblk, 0, stream>>>(apack, wpack + (size_t)(l * 5 + 4) * WMAT,
                                               mlp_b + (size_t)l * H, hbuf, Mtiles, H, 0);
        ln_relu<<<NN / 8, gblk, 0, stream>>>(hbuf, ln_g + (size_t)l * H, ln_b + (size_t)l * H, xa);
        cur = xa;
    }

    // ---- heads ----
    pack_a<<<PA, gblk, 0, stream>>>(xa, apack, Mtiles);
    gemm_bf16<<<ggemmO, gblk, 0, stream>>>(apack, wp_lin, lin_b, out, Mtiles, OUTC, 0);                    // mu
    gemm_bf16<<<ggemmH, gblk, 0, stream>>>(apack, wp_f1,  f1_b,  tbuf, Mtiles, H, 1);                      // relu(f1)
    pack_a<<<PA, gblk, 0, stream>>>(tbuf, apack, Mtiles);
    gemm_bf16<<<ggemmO, gblk, 0, stream>>>(apack, wp_f2,  f2_b,  out + (size_t)NN * OUTC, Mtiles, OUTC, 0); // logvar
}
